// LinearCrossEntropyLoss_19361712570791
// MI455X (gfx1250) — 8-bit matrix operand order repaired and checked by reading, compiled, not run on hardware
//
#include <hip/hip_runtime.h>
#include <math.h>


typedef __attribute__((ext_vector_type(16))) int   v16i;
typedef __attribute__((ext_vector_type(8)))  float v8f;

#define DDIM 2048
#define VOCAB 50257
#define NROWS 4096
#define KC 128
#define NKC (DDIM / KC)
#define BV 128
#define BM 128
#define VBLOCKS ((VOCAB + BV - 1) / BV)
#define NRB (NROWS / BM)
#define NPART (VBLOCKS * 2)
#define IGNORE_INDEX (-100)
#define Z_REG 1e-4f
#define LOGIT_SCALE 1.0f
#define LDS_TILE (BV * 144)

#define XQ_OFF   ((size_t)0)
#define XQ_SZ    ((size_t)NROWS * DDIM)
#define WQ_OFF   (XQ_OFF + XQ_SZ)
#define WQ_SZ    ((size_t)VOCAB * DDIM)
#define PART_OFF ((WQ_OFF + WQ_SZ + 255) & ~(size_t)255)
#define PART_SZ  ((size_t)NPART * NROWS * sizeof(float2))
#define LSE_OFF  ((PART_OFF + PART_SZ + 255) & ~(size_t)255)
#define PICK_OFF (LSE_OFF + (size_t)NROWS * sizeof(float) + 256)

__device__ __forceinline__ unsigned int f32_to_e4m3(float f) {
  unsigned int u = __float_as_uint(f);
  unsigned int sign = (u >> 24) & 0x80u;
  unsigned int a = u & 0x7FFFFFFFu;
  if (a >= 0x43E00000u) {
    return sign | (a > 0x7F800000u ? 0x7Fu : 0x7Eu);
  }
  if (a < 0x3C800000u) {
    float s = __uint_as_float(a) * 512.0f;
    unsigned int m = (unsigned int)(s + 0.5f);
    if (m > 7u) return sign | 0x08u;
    return sign | m;
  }
  unsigned int e    = (a >> 23) - 120u;
  unsigned int mant = (a >> 20) & 0x7u;
  unsigned int rest = a & 0xFFFFFu;
  unsigned int rnd  = (rest > 0x80000u) || (rest == 0x80000u && (mant & 1u));
  unsigned int code = (e << 3) | mant;
  code += rnd;
  if (code > 0x7Eu) code = 0x7Eu;
  return sign | code;
}

__device__ __forceinline__ unsigned int pack4_e4m3(float4 f) {
  return f32_to_e4m3(f.x) | (f32_to_e4m3(f.y) << 8) |
         (f32_to_e4m3(f.z) << 16) | (f32_to_e4m3(f.w) << 24);
}

__global__ __launch_bounds__(256) void convert_x_pack(
    const float* __restrict__ x, unsigned char* __restrict__ xq) {
  size_t idx = (size_t)blockIdx.x * 256 + threadIdx.x;
  if (idx >= (size_t)NROWS * DDIM / 4) return;
  int row = (int)(idx >> 9);
  int k   = ((int)idx & 511) * 4;
  float4 f = ((const float4*)x)[idx];
  unsigned int p = pack4_e4m3(f);
  int rt = row >> 4, rl = row & 15;
  int kc = k >> 7,  kp = k & 127;
  int sub = kp >> 6, kk = kp & 63;
  int h = (kk >> 4) & 1, c = kk >> 5, j = kk & 15;
  int v = sub * 8 + c * 4 + (j >> 2);
  int lane = rl + 16 * h;
  size_t dst = (((size_t)rt * NKC + kc) << 11) + (size_t)(lane * 64 + v * 4);
  *(unsigned int*)(xq + dst) = p;
}

__global__ __launch_bounds__(256) void convert_w(
    const float* __restrict__ w, unsigned char* __restrict__ wq) {
  size_t idx = (size_t)blockIdx.x * 256 + threadIdx.x;
  if (idx >= (size_t)VOCAB * DDIM / 4) return;
  float4 f = ((const float4*)w)[idx];
  ((unsigned int*)wq)[idx] = pack4_e4m3(f);
}

__global__ __launch_bounds__(256) void gemm_partial(
    const unsigned char* __restrict__ xq,
    const unsigned char* __restrict__ wq,
    float2* __restrict__ part) {
  __shared__ int4 sWtile[(2 * LDS_TILE) / 16];
  unsigned char* sWb = (unsigned char*)sWtile;
  const unsigned lbase = (unsigned)(uintptr_t)sWb;

  const int tid  = threadIdx.x;
  const int wave = tid >> 5;
  const int lane = tid & 31;
  const int wm = wave & 3;
  const int wn = wave >> 2;
  const int l16 = lane & 15;
  const int half = lane >> 4;
  const int vb = blockIdx.x;
  const int rb = blockIdx.y;
  const int colbase = vb * BV;

  const unsigned char* gp[4];
  unsigned sdst[4];
#pragma unroll
  for (int i = 0; i < 4; ++i) {
    int f = tid + i * 256;
    int c = f >> 3, j = f & 7;
    int gc = colbase + c;
    if (gc >= VOCAB) gc = VOCAB - 1;
    gp[i] = wq + (size_t)gc * DDIM + (size_t)(j * 16);
    sdst[i] = (unsigned)(c * 144 + j * 16);
  }

  v8f acc[2][4] = {};

  const int rt0 = rb * 8 + wm * 2;
  const unsigned char* ap =
      xq + (((size_t)rt0 * NKC) << 11) + (size_t)(lane << 6);

#pragma unroll
  for (int i = 0; i < 4; ++i) {
    unsigned laddr = lbase + sdst[i];
    asm volatile("global_load_async_to_lds_b128 %0, %1, off"
                 :: "v"(laddr), "v"(gp[i]) : "memory");
  }
#pragma unroll
  for (int i = 0; i < 4; ++i) gp[i] += KC;

#pragma unroll 1
  for (int kc = 0; kc < NKC; ++kc) {
    const int buf = kc & 1;
    if (kc + 1 < NKC) {
      const unsigned boff = (unsigned)((buf ^ 1) * LDS_TILE);
#pragma unroll
      for (int i = 0; i < 4; ++i) {
        unsigned laddr = lbase + boff + sdst[i];
        asm volatile("global_load_async_to_lds_b128 %0, %1, off"
                     :: "v"(laddr), "v"(gp[i]) : "memory");
      }
#pragma unroll
      for (int i = 0; i < 4; ++i) gp[i] += KC;
    }

    union U16 { v16i v; int4 q[4]; };
    U16 Ua[2];
#pragma unroll
    for (int mi = 0; mi < 2; ++mi) {
      const int4* a4 = (const int4*)(ap + (size_t)mi * ((size_t)NKC << 11));
      Ua[mi].q[0] = a4[0]; Ua[mi].q[1] = a4[1];
      Ua[mi].q[2] = a4[2]; Ua[mi].q[3] = a4[3];
    }
    ap += (size_t)1 << 11;

    if (kc + 1 < NKC) asm volatile("s_wait_asynccnt 0x4" ::: "memory");
    else              asm volatile("s_wait_asynccnt 0x0" ::: "memory");
    __syncthreads();

    const int bufoff = buf * LDS_TILE;
    U16 Ub[2];
    {
      int bbase = bufoff + (wn * 64 + l16) * 144 + (half << 4);
      Ub[0].q[0] = *(const int4*)(sWb + bbase);
      Ub[0].q[1] = *(const int4*)(sWb + bbase + 32);
      Ub[0].q[2] = *(const int4*)(sWb + bbase + 64);
      Ub[0].q[3] = *(const int4*)(sWb + bbase + 96);
    }
#pragma unroll
    for (int ni = 0; ni < 4; ++ni) {
      if (ni + 1 < 4) {
        int bbase = bufoff + (wn * 64 + (ni + 1) * 16 + l16) * 144 + (half << 4);
        int slot = (ni + 1) & 1;
        Ub[slot].q[0] = *(const int4*)(sWb + bbase);
        Ub[slot].q[1] = *(const int4*)(sWb + bbase + 32);
        Ub[slot].q[2] = *(const int4*)(sWb + bbase + 64);
        Ub[slot].q[3] = *(const int4*)(sWb + bbase + 96);
      }
#pragma unroll
      for (int mi = 0; mi < 2; ++mi) {
        acc[mi][ni] = __builtin_amdgcn_wmma_f32_16x16x128_fp8_fp8(
            Ua[mi].v, Ub[ni & 1].v, (short)0, acc[mi][ni], false, false);
      }
    }
    __syncthreads();
  }

  const int row0 = rb * BM + wm * 32;
#pragma unroll
  for (int mi = 0; mi < 2; ++mi) {
#pragma unroll
    for (int r = 0; r < 8; ++r) {
      float vals[4];
      float m = -INFINITY;
#pragma unroll
      for (int ni = 0; ni < 4; ++ni) {
        int col = colbase + wn * 64 + ni * 16 + l16;
        float t = acc[mi][ni][r] * LOGIT_SCALE;
        t = (col < VOCAB) ? t : -INFINITY;
        vals[ni] = t;
        m = fmaxf(m, t);
      }
      m = fmaxf(m, __shfl_xor(m, 1, 32));
      m = fmaxf(m, __shfl_xor(m, 2, 32));
      m = fmaxf(m, __shfl_xor(m, 4, 32));
      m = fmaxf(m, __shfl_xor(m, 8, 32));
      float s = 0.f;
#pragma unroll
      for (int ni = 0; ni < 4; ++ni)
        if (vals[ni] > -INFINITY) s += __expf(vals[ni] - m);
      s += __shfl_xor(s, 1, 32);
      s += __shfl_xor(s, 2, 32);
      s += __shfl_xor(s, 4, 32);
      s += __shfl_xor(s, 8, 32);
      if (l16 == 0) {
        int row = row0 + mi * 16 + half * 8 + r;
        part[(size_t)(vb * 2 + wn) * NROWS + row] = make_float2(m, s);
      }
    }
  }
}

__global__ __launch_bounds__(256) void reduce_lse(
    const float2* __restrict__ part, float* __restrict__ lse) {
  const int row = blockIdx.x;
  const int t = threadIdx.x;
  float m = -INFINITY, s = 0.f;
  for (int i = t; i < NPART; i += 256) {
    float2 p = part[(size_t)i * NROWS + row];
    float M = fmaxf(m, p.x);
    float S = 0.f;
    if (s > 0.f)   S += s   * __expf(m   - M);
    if (p.y > 0.f) S += p.y * __expf(p.x - M);
    m = M; s = S;
  }
  __shared__ float sm[256], ss[256];
  sm[t] = m; ss[t] = s;
  __syncthreads();
  for (int off = 128; off > 0; off >>= 1) {
    if (t < off) {
      float m2 = sm[t + off], s2 = ss[t + off];
      float M = fmaxf(sm[t], m2);
      float S = 0.f;
      if (ss[t] > 0.f) S += ss[t] * __expf(sm[t] - M);
      if (s2 > 0.f)    S += s2    * __expf(m2    - M);
      sm[t] = M; ss[t] = S;
    }
    __syncthreads();
  }
  if (t == 0) lse[row] = (ss[0] > 0.f) ? sm[0] + __logf(ss[0]) : -INFINITY;
}

__global__ __launch_bounds__(256) void picked_kernel(
    const float* __restrict__ x, const int* __restrict__ labels,
    const float* __restrict__ w, float* __restrict__ picked) {
  const int wv = threadIdx.x >> 5;
  const int lane = threadIdx.x & 31;
  const int row = blockIdx.x * 8 + wv;
  if (row >= NROWS) return;
  int lab = labels[row];
  if (lab < 0 || lab >= VOCAB) {
    if (lane == 0) picked[row] = 0.f;
    return;
  }
  const float4* xr = (const float4*)(x + (size_t)row * DDIM);
  const float4* wr = (const float4*)(w + (size_t)lab * DDIM);
  float acc = 0.f;
  for (int i = lane; i < DDIM / 4; i += 32) {
    float4 a = xr[i], b = wr[i];
    acc += a.x * b.x + a.y * b.y + a.z * b.z + a.w * b.w;
  }
  acc += __shfl_xor(acc, 16, 32);
  acc += __shfl_xor(acc, 8, 32);
  acc += __shfl_xor(acc, 4, 32);
  acc += __shfl_xor(acc, 2, 32);
  acc += __shfl_xor(acc, 1, 32);
  if (lane == 0) picked[row] = acc * LOGIT_SCALE;
}

__global__ __launch_bounds__(256) void final_kernel(
    const int* __restrict__ labels, const float* __restrict__ lse,
    const float* __restrict__ picked, float* __restrict__ out) {
  const int t = threadIdx.x;
  float nll = 0.f, zz = 0.f, cnt = 0.f;
  for (int i = t; i < NROWS; i += 256) {
    int lab = labels[i];
    if (lab != IGNORE_INDEX) {
      float l = lse[i];
      nll += l - picked[i];
      zz  += l * l;
      cnt += 1.f;
    }
  }
  __shared__ float sa[256], sb[256], sc[256];
  sa[t] = nll; sb[t] = zz; sc[t] = cnt;
  __syncthreads();
  for (int off = 128; off > 0; off >>= 1) {
    if (t < off) {
      sa[t] += sa[t + off];
      sb[t] += sb[t + off];
      sc[t] += sc[t + off];
    }
    __syncthreads();
  }
  if (t == 0) {
    float denom = fmaxf(sc[0], 1.f);
    float loss = sa[0] / denom;
    if (sc[0] > 0.f) loss += Z_REG * (sb[0] / denom);
    out[0] = loss;
  }
}

extern "C" void kernel_launch(void* const* d_in, const int* in_sizes, int n_in,
                              void* d_out, int out_size, void* d_ws,
                              size_t ws_size, hipStream_t stream) {
  const float* x      = (const float*)d_in[0];
  const int*   labels = (const int*)d_in[1];
  const float* weight = (const float*)d_in[2];

  unsigned char* ws = (unsigned char*)d_ws;
  unsigned char* xq = ws + XQ_OFF;
  unsigned char* wq = ws + WQ_OFF;
  float2* part = (float2*)(ws + PART_OFF);
  float*  lse  = (float*)(ws + LSE_OFF);
  float*  pick = (float*)(ws + PICK_OFF);

  {
    int blocks = (int)(((size_t)NROWS * DDIM / 4 + 255) / 256);
    convert_x_pack<<<blocks, 256, 0, stream>>>(x, xq);
  }
  {
    int blocks = (int)(((size_t)VOCAB * DDIM / 4 + 255) / 256);
    convert_w<<<blocks, 256, 0, stream>>>(weight, wq);
  }
  gemm_partial<<<dim3(VBLOCKS, NRB), 256, 0, stream>>>(xq, wq, part);
  reduce_lse<<<NROWS, 256, 0, stream>>>(part, lse);
  picked_kernel<<<NROWS / 8, 256, 0, stream>>>(x, labels, weight, pick);
  final_kernel<<<1, 256, 0, stream>>>(labels, lse, pick, (float*)d_out);
}
